// STDCN_with_GRU_37580963840609
// MI455X (gfx1250) — matrix operand order repaired and checked by reading, not compiled, not run on hardware
//
#include <hip/hip_runtime.h>
#include <hip/hip_bf16.h>
#include <stdint.h>

#define B_SZ 1024
#define N_SZ 64
#define T_SZ 256
#define HID  128
#define G3   (3 * HID)

typedef __attribute__((ext_vector_type(16))) __bf16 v16bf;
typedef __attribute__((ext_vector_type(8)))  float  v8f;
typedef __attribute__((ext_vector_type(4)))  unsigned int u32x4;
typedef __attribute__((ext_vector_type(8)))  int i32x8;
typedef __attribute__((ext_vector_type(4)))  int i32x4;

#if defined(__has_builtin)
#  if __has_builtin(__builtin_amdgcn_tensor_load_to_lds)
#    define HAVE_TDM 1
#  endif
#endif
#if defined(HAVE_TDM)
#  if __has_include(<hip/amd_detail/amd_gfx1250_TDM.h>)
#    define TDM_6ARG 1
#  endif
#endif

union Frag { v16bf v; unsigned u[8]; };

static __device__ __forceinline__ unsigned short f32_to_bf16(float f) {
  union { float f; unsigned u; } x; x.f = f;
  unsigned r = x.u + 0x7FFFu + ((x.u >> 16) & 1u);
  return (unsigned short)(r >> 16);
}

static __device__ __forceinline__ float sigmoidf_(float x) {
  return 1.0f / (1.0f + __expf(-x));
}

static __device__ __forceinline__ v8f wmma_bf16(const Frag& a, const Frag& b, v8f c) {
  return __builtin_amdgcn_wmma_f32_16x16x32_bf16(
      /*neg_a=*/false, a.v, /*neg_b=*/false, b.v,
      /*c_mod=*/(short)0, c, /*reuse_a=*/false, /*reuse_b=*/false);
}

// A-matrix bf16 16x32 fragment: lane<16 -> K in {0..15},
// lane>=16 -> K in {16..31}; K-pairs are contiguous -> single u32 loads.
static __device__ __forceinline__ void load_a16(Frag& f, const unsigned short* p0,
                                                int ldRow, int k0) {
  const int lane = threadIdx.x & 31;
  const int m  = lane & 15;
  const int kh = (lane >> 4) << 4;  // 0 or 16
  const unsigned short* p = p0 + (size_t)m * ldRow + k0 + kh;
#pragma unroll
  for (int g = 0; g < 8; ++g) {
    const int k = 2 * g;  // 0,2,4,...,14
    unsigned v; __builtin_memcpy(&v, p + k, 4);
    f.u[g] = v;
  }
}

// B-matrix bf16 32x16 fragment from storage w[col][k] (k contiguous, f32 src).
static __device__ __forceinline__ void load_b16_kcontig_f32(Frag& f, const float* w,
                                                            int ldCol, int col0, int k0) {
  const int lane = threadIdx.x & 31;
  const int col = col0 + (lane & 15);
  const int kh  = (lane >> 4) << 4;  // 0 or 16
  const float* p = w + (size_t)col * ldCol + k0 + kh;
#pragma unroll
  for (int g = 0; g < 8; ++g)
    f.u[g] = (unsigned)f32_to_bf16(p[2 * g]) |
             ((unsigned)f32_to_bf16(p[2 * g + 1]) << 16);
}

// B-matrix bf16 32x16 fragment from bf16 storage w[col][k] (k contiguous).
static __device__ __forceinline__ void load_b16_kcontig_u16(Frag& f,
                                                            const unsigned short* w,
                                                            int ldCol, int col0, int k0) {
  const int lane = threadIdx.x & 31;
  const int col = col0 + (lane & 15);
  const int kh  = (lane >> 4) << 4;  // 0 or 16
  const unsigned short* p = w + (size_t)col * ldCol + k0 + kh;
#pragma unroll
  for (int g = 0; g < 8; ++g) {
    unsigned v; __builtin_memcpy(&v, p + 2 * g, 4);
    f.u[g] = v;
  }
}

// ---------------------------------------------------------------------------
// Kernel 1: normalized adjacency + GCN conv + scale + leaky-relu, output
// transposed bf16 x[B][T][N]. One block per batch; X[b] (64 KB, contiguous)
// is DMA'd into LDS by the Tensor Data Mover where available.
// ---------------------------------------------------------------------------
__global__ __launch_bounds__(256) void gcn_kernel(const float* __restrict__ X,
                                                  const float* __restrict__ H,
                                                  const float* __restrict__ wscal,
                                                  unsigned short* __restrict__ xT) {
  __shared__ float Hs[N_SZ * N_SZ];             // 16 KB
  __shared__ float dinv[N_SZ];
  __shared__ unsigned short Hbf[N_SZ * N_SZ];   // 8 KB   (A operand, row-major)
  __shared__ unsigned short Xbf[T_SZ * N_SZ];   // 32 KB  (B operand, [t][node], K-contig)
#if defined(HAVE_TDM)
  __shared__ float Xstage[N_SZ * T_SZ];         // 64 KB TDM landing buffer
#endif

  const int b = blockIdx.x;
  const int tid = threadIdx.x;
  const float wmul = wscal[0];
  const float* Xb = X + (size_t)b * N_SZ * T_SZ;

#if defined(HAVE_TDM)
  if (tid < 32) {  // wave 0 issues the tensor DMA (descriptor is uniform)
    u32x4 g0 = {};
    i32x8 g1 = {};
    i32x4 g2 = {}, g3 = {};
    const unsigned long long ga = (unsigned long long)(const void*)Xb;
    g0[0] = 1u;                                        // count=1, user mode
    g0[1] = (unsigned)(unsigned long long)(const void*)Xstage;   // lds_addr
    g0[2] = (unsigned)ga;                              // global_addr[31:0]
    g0[3] = (unsigned)((ga >> 32) & 0x01FFFFFFull) | (2u << 30); // addr[56:32]|type=2
    g1[0] = (int)0x00020000;  // data_size = 4 bytes
    g1[1] = (int)0x40000000;  // tensor_dim0 = 16384 (low16 -> bits[31:16])
    g1[2] = (int)0x00010000;  // tensor_dim0 hi = 0 | tensor_dim1 = 1
    g1[3] = (int)0x40000000;  // tensor_dim1 hi = 0 | tile_dim0 = 16384
    g1[4] = 1;                // tile_dim1 = 1, tile_dim2 = 0
    g1[5] = 16384;            // tensor_dim0_stride (low 32)
#if defined(TDM_6ARG)
    i32x8 g4 = {};
    __builtin_amdgcn_tensor_load_to_lds(g0, g1, g2, g3, g4, 0);
#else
    __builtin_amdgcn_tensor_load_to_lds(g0, g1, g2, g3, 0);
#endif
    __builtin_amdgcn_s_wait_tensorcnt(0);
  }
#endif

  for (int i = tid; i < N_SZ * N_SZ; i += 256) Hs[i] = H[(size_t)b * N_SZ * N_SZ + i];
  __syncthreads();

  if (tid < N_SZ) {
    float s = 0.0f;
    for (int j = 0; j < N_SZ; ++j) s += Hs[tid * N_SZ + j];
    s = s - Hs[tid * N_SZ + tid] + 1.0f;   // zero diag, add identity
    dinv[tid] = rsqrtf(s);
  }
  __syncthreads();

  for (int e = tid; e < N_SZ * N_SZ; e += 256) {
    const int i = e >> 6, j = e & 63;
    const float v = (i == j) ? 1.0f : Hs[e];
    Hbf[e] = f32_to_bf16(v * dinv[i] * dinv[j]);
  }
  // X[b] row-major [node][t]  ->  Xbf col-major [t][node] bf16 (K-contig B feed)
  for (int e = tid; e < N_SZ * T_SZ; e += 256) {
    const int n = e >> 8, t = e & 255;
#if defined(HAVE_TDM)
    Xbf[t * N_SZ + n] = f32_to_bf16(Xstage[e]);
#else
    Xbf[t * N_SZ + n] = f32_to_bf16(Xb[e]);
#endif
  }
  __syncthreads();

  const int wave = tid >> 5;
  const int lane = tid & 31;
  const int mt = (wave & 3) << 4;   // this wave's (constant) node-tile base
  const int q  = wave >> 2;

  Frag A0, A1;  // Hn fragments: loaded once, reused for 8 time-tiles
  load_a16(A0, Hbf + mt * N_SZ, N_SZ, 0);
  load_a16(A1, Hbf + mt * N_SZ, N_SZ, 32);

#pragma unroll
  for (int i = 0; i < 8; ++i) {
    const int nt = (q + 2 * i) << 4;  // time-tile base
    Frag b0, b1;
    load_b16_kcontig_u16(b0, Xbf, N_SZ, nt, 0);
    load_b16_kcontig_u16(b1, Xbf, N_SZ, nt, 32);
    v8f acc = {};
    acc = wmma_bf16(A0, b0, acc);
    acc = wmma_bf16(A1, b1, acc);

    // C layout: VGPR g -> M = g + 8*(lane>=16); N = lane&15
    const int tcol  = nt + (lane & 15);
    const int mbase = mt + ((lane >> 4) << 3);
    unsigned short* dst = xT + ((size_t)b * T_SZ + tcol) * N_SZ + mbase;
#pragma unroll
    for (int g = 0; g < 8; ++g) {
      float v = acc[g] * wmul;
      v = v > 0.0f ? v : 0.01f * v;  // leaky relu
      dst[g] = f32_to_bf16(v);
    }
  }
}

// ---------------------------------------------------------------------------
// Kernel 2: fused bidirectional GRU. grid = (B/16, 2), 256 threads.
// Wave w owns hidden columns [16w,16w+16); all three gates stay in-wave.
// Weight B-fragments loaded once into VGPRs for all 256 steps; h state in
// registers; ping-pong 4 KB LDS bf16 mirror -> one barrier per step.
// ---------------------------------------------------------------------------
__global__ __launch_bounds__(256) void gru_kernel(
    const unsigned short* __restrict__ xT,
    const float* __restrict__ wih_f, const float* __restrict__ whh_f,
    const float* __restrict__ bih_f, const float* __restrict__ bhh_f,
    const float* __restrict__ wih_b, const float* __restrict__ whh_b,
    const float* __restrict__ bih_b, const float* __restrict__ bhh_b,
    float* __restrict__ out) {
  __shared__ unsigned short hbuf[2][16 * HID];  // 8 KB ping-pong

  const int dir  = blockIdx.y;
  const int b0   = blockIdx.x << 4;
  const int tid  = threadIdx.x;
  const int wave = tid >> 5;
  const int lane = tid & 31;
  const int j0   = wave << 4;
  const int jc   = j0 + (lane & 15);

  const float* wih = dir ? wih_b : wih_f;
  const float* whh = dir ? whh_b : whh_f;
  const float* bih = dir ? bih_b : bih_f;
  const float* bhh = dir ? bhh_b : bhh_f;

  // Register-resident weight fragments (loaded once).
  Frag Bh[3][4], Bx[3][2];
#pragma unroll
  for (int gte = 0; gte < 3; ++gte) {
#pragma unroll
    for (int kk = 0; kk < 4; ++kk)
      load_b16_kcontig_f32(Bh[gte][kk], whh, HID, gte * HID + j0, kk * 32);
#pragma unroll
    for (int kk = 0; kk < 2; ++kk)
      load_b16_kcontig_f32(Bx[gte][kk], wih, N_SZ, gte * HID + j0, kk * 32);
  }
  const float brz0 = bih[jc] + bhh[jc];
  const float brz1 = bih[HID + jc] + bhh[HID + jc];
  const float bxn  = bih[2 * HID + jc];
  const float bhn  = bhh[2 * HID + jc];

  for (int e = tid; e < 16 * HID; e += 256) hbuf[0][e] = 0;
  float hreg[8];
#pragma unroll
  for (int g = 0; g < 8; ++g) hreg[g] = 0.0f;
  __syncthreads();

  for (int t = 0; t < T_SZ; ++t) {
    const int tt = dir ? (T_SZ - 1 - t) : t;
    const unsigned short* hsrc = hbuf[t & 1];
    unsigned short* hdst = hbuf[(t & 1) ^ 1];

    Frag Ax[2], Ah[4];
    const unsigned short* xbase = xT + ((size_t)b0 * T_SZ + tt) * N_SZ;
#pragma unroll
    for (int kk = 0; kk < 2; ++kk) load_a16(Ax[kk], xbase, T_SZ * N_SZ, kk * 32);
#pragma unroll
    for (int kk = 0; kk < 4; ++kk) load_a16(Ah[kk], hsrc, HID, kk * 32);

    if (t + 1 < T_SZ) {  // prefetch next x_t (global_prefetch_b8)
      const int tn = dir ? (tt - 1) : (tt + 1);
      __builtin_prefetch(xT + ((size_t)b0 * T_SZ + tn) * N_SZ + (lane << 2), 0, 3);
    }

    v8f Sx[3], Sh[3];
#pragma unroll
    for (int gte = 0; gte < 3; ++gte) {
      v8f ax = {};
      ax = wmma_bf16(Ax[0], Bx[gte][0], ax);
      ax = wmma_bf16(Ax[1], Bx[gte][1], ax);
      Sx[gte] = ax;
      v8f ah = {};
#pragma unroll
      for (int kk = 0; kk < 4; ++kk) ah = wmma_bf16(Ah[kk], Bh[gte][kk], ah);
      Sh[gte] = ah;
    }

#pragma unroll
    for (int g = 0; g < 8; ++g) {
      const int m = g + ((lane >> 4) << 3);  // batch row within block
      const float r = sigmoidf_(Sx[0][g] + Sh[0][g] + brz0);
      const float z = sigmoidf_(Sx[1][g] + Sh[1][g] + brz1);
      const float n = tanhf((Sx[2][g] + bxn) + r * (Sh[2][g] + bhn));
      const float h = (1.0f - z) * n + z * hreg[g];
      hreg[g] = h;
      hdst[m * HID + jc] = f32_to_bf16(h);
      out[((size_t)(b0 + m) * (2 * HID) + (size_t)(dir * HID + jc)) * T_SZ + tt] = h;
    }
    __syncthreads();  // hdst complete & hsrc reads done -> next step
  }
}

// ---------------------------------------------------------------------------
// BatchNorm1d(T) in training mode: stats over (B, 2H) per t, then normalize.
// ---------------------------------------------------------------------------
__global__ void bn_init_kernel(float* __restrict__ stats) {
  const int i = blockIdx.x * blockDim.x + threadIdx.x;
  if (i < 2 * T_SZ) stats[i] = 0.0f;
}

__global__ __launch_bounds__(256) void bn_stats_kernel(const float* __restrict__ y,
                                                       float* __restrict__ stats) {
  const int t = threadIdx.x;                 // thread owns one t
  const size_t r0 = (size_t)blockIdx.x * 64; // 64 (b,c) rows per block
  float s1 = 0.0f, s2 = 0.0f;
  for (int r = 0; r < 64; ++r) {
    const float v = y[(r0 + r) * T_SZ + t];
    s1 += v; s2 += v * v;
  }
  atomicAdd(&stats[t], s1);
  atomicAdd(&stats[T_SZ + t], s2);
}

__global__ __launch_bounds__(256) void bn_norm_kernel(float* __restrict__ y,
                                                      const float* __restrict__ stats,
                                                      const float* __restrict__ gamma,
                                                      const float* __restrict__ beta) {
  const float inv = 1.0f / (float)((size_t)B_SZ * 2 * HID);
  const size_t total = (size_t)B_SZ * 2 * HID * T_SZ;
  for (size_t idx = (size_t)blockIdx.x * blockDim.x + threadIdx.x; idx < total;
       idx += (size_t)gridDim.x * blockDim.x) {
    const int t = (int)(idx & (size_t)(T_SZ - 1));
    const float mean = stats[t] * inv;
    const float var  = stats[T_SZ + t] * inv - mean * mean;
    y[idx] = (y[idx] - mean) * rsqrtf(var + 1e-5f) * gamma[t] + beta[t];
  }
}

extern "C" void kernel_launch(void* const* d_in, const int* in_sizes, int n_in,
                              void* d_out, int out_size, void* d_ws, size_t ws_size,
                              hipStream_t stream) {
  const float* X     = (const float*)d_in[0];
  const float* H     = (const float*)d_in[1];
  const float* wscal = (const float*)d_in[2];
  const float* wih_f = (const float*)d_in[3];
  const float* whh_f = (const float*)d_in[4];
  const float* bih_f = (const float*)d_in[5];
  const float* bhh_f = (const float*)d_in[6];
  const float* wih_b = (const float*)d_in[7];
  const float* whh_b = (const float*)d_in[8];
  const float* bih_b = (const float*)d_in[9];
  const float* bhh_b = (const float*)d_in[10];
  const float* gamma = (const float*)d_in[11];
  const float* beta  = (const float*)d_in[12];
  float* out = (float*)d_out;

  unsigned short* xT = (unsigned short*)d_ws;  // bf16 x[B][T][N], 33.5 MB
  float* stats = (float*)((char*)d_ws +
                          (size_t)B_SZ * T_SZ * N_SZ * sizeof(unsigned short));

  gcn_kernel<<<dim3(B_SZ), dim3(256), 0, stream>>>(X, H, wscal, xT);
  gru_kernel<<<dim3(B_SZ / 16, 2), dim3(256), 0, stream>>>(
      xT, wih_f, whh_f, bih_f, bhh_f, wih_b, whh_b, bih_b, bhh_b, out);
  bn_init_kernel<<<dim3(2), dim3(256), 0, stream>>>(stats);
  bn_stats_kernel<<<dim3((B_SZ * 2 * HID) / 64), dim3(256), 0, stream>>>(out, stats);
  bn_norm_kernel<<<dim3(4096), dim3(256), 0, stream>>>(out, stats, gamma, beta);
}
